// PriorModule_42502996361294
// MI455X (gfx1250) — compile-verified, matrix operand order checked by reading, not run on hardware
//
#include <hip/hip_runtime.h>
#include <hip/hip_bf16.h>

// ---------------------------------------------------------------------------
// MI455X (gfx1250) fused autoregressive prior.
//
// KV-cache everything; bf16 WMMA (v_wmma_f32_16x16x32_bf16) with fp32
// accumulate for all GEMMs; fp32 VALU softmax/LN/exp. One fused kernel per
// step; each WG owns 16 batch rows (every phase is row-local -> only
// __syncthreads between phases). Cross-attn K2/V2 bf16 caches (134 MB) are
// L2-resident (192 MB L2). GEMM inner loop is double-buffered so WMMAs issue
// back-to-back while next fragments are in flight. Tile staging uses
// GLOBAL_LOAD_ASYNC_TO_LDS_B128 (ASYNCcnt) instead of a VGPR round trip.
// ---------------------------------------------------------------------------

typedef __attribute__((ext_vector_type(16))) __bf16 v16bf;
typedef __attribute__((ext_vector_type(8)))  __bf16 v8bf;
typedef __attribute__((ext_vector_type(8)))  float  v8f;

#define D_MODEL 512
#define NHEADS  8
#define DHEAD   64
#define DLAT    64
#define TT      32
#define BB      512
#define LC      128
#define MT      16      // batch rows per workgroup (WMMA M tile)
#define NTHREADS 256    // 8 wave32 waves

// ---------------------------------------------------------------------------
// Fragment-layout weight packing (ISA 7.12.2, 16-bit A/B 16x32 / 32x16):
//   frag element e of lane L covers k = koff + (e<8 ? e : e+8), koff=(L>>4)*8
//   packed index: ((nt*KC + kc)*32 + lane)*16 + e  ->  W[k_global][n_global]
// ---------------------------------------------------------------------------
__global__ void pack_w_kernel(const float* __restrict__ src, __bf16* __restrict__ dst,
                              int KC, int N) {
    int t = blockIdx.x * blockDim.x + threadIdx.x;
    int total = KC * 32 * N;            // == K*N
    if (t >= total) return;
    int e    = t & 15;
    int lane = (t >> 4) & 31;
    int rest = t >> 9;
    int kc   = rest % KC;
    int nt   = rest / KC;
    int n = nt * 16 + (lane & 15);
    int k = kc * 32 + ((lane >> 4) << 3) + e + ((e >= 8) ? 8 : 0);
    dst[t] = (__bf16)src[(size_t)k * N + n];
}

__global__ void init_x_kernel(const float* __restrict__ w0, float* __restrict__ xBuf) {
    int t = blockIdx.x * blockDim.x + threadIdx.x;
    if (t < BB * D_MODEL) xBuf[t] = w0[t & (D_MODEL - 1)];
}

// ---------------------------------------------------------------------------
// Async global->LDS tile staging (gfx1250 GLOBAL_LOAD_ASYNC_TO_LDS_B128, GVS
// mode: mem = SADDR + VGPR_offset). Per ISA 10.2, the low 32 bits of a
// generic pointer into the LDS aperture are the LDS byte address. Tracked by
// ASYNCcnt; drained with s_wait_asynccnt before the block barrier.
// ---------------------------------------------------------------------------
__device__ __forceinline__ void async_tile_load_f32(float* ldsDst,
                                                    const float* __restrict__ gSrc,
                                                    int nFloats, int tid) {
    unsigned ldsBase = (unsigned)(size_t)ldsDst;
#pragma unroll 2
    for (int idx = tid * 4; idx < nFloats; idx += NTHREADS * 4) {
        unsigned lds  = ldsBase + (unsigned)idx * 4u;
        unsigned voff = (unsigned)idx * 4u;
        asm volatile("global_load_async_to_lds_b128 %0, %1, %2"
                     :: "v"(lds), "v"(voff), "s"(gSrc)
                     : "memory");
    }
    asm volatile("s_wait_asynccnt 0x0" ::: "memory");
}

__device__ __forceinline__ v16bf load_afrag(const __bf16* aRow, int kc) {
    v8bf lo = *(const v8bf*)(aRow + kc * 32);
    v8bf hi = *(const v8bf*)(aRow + kc * 32 + 16);
    v16bf a;
#pragma unroll
    for (int e = 0; e < 8; ++e) { a[e] = lo[e]; a[8 + e] = hi[e]; }
    return a;
}

// ---------------------------------------------------------------------------
// One wave computes a 16 x (NT*16) tile: A from LDS (row-major bf16, stride
// AS), B from fragment-packed weights in global memory. Double-buffered so
// the 4 WMMAs of iteration kc overlap the loads of iteration kc+1.
// ---------------------------------------------------------------------------
template<int KC, int NT>
__device__ __forceinline__ void wave_gemm(const __bf16* sA, int AS,
                                          const __bf16* __restrict__ pkW,
                                          int ntBase, v8f* acc) {
    const int lane = threadIdx.x & 31;
    const int row  = lane & 15;
    const int koff = (lane >> 4) << 3;
    const __bf16* aRow  = sA + row * AS + koff;
    const __bf16* bBase = pkW + (size_t)ntBase * KC * 512 + lane * 16;

    v16bf a = load_afrag(aRow, 0);
    v16bf b[NT];
#pragma unroll
    for (int t = 0; t < NT; ++t)
        b[t] = *(const v16bf*)(bBase + (size_t)t * KC * 512);

#pragma unroll
    for (int kc = 0; kc < KC; ++kc) {
        v16bf aN, bN[NT];
        const bool more = (kc + 1 < KC);
        if (more) {
            aN = load_afrag(aRow, kc + 1);
#pragma unroll
            for (int t = 0; t < NT; ++t)
                bN[t] = *(const v16bf*)(bBase + ((size_t)t * KC + (kc + 1)) * 512);
        }
#pragma unroll
        for (int t = 0; t < NT; ++t)
            acc[t] = __builtin_amdgcn_wmma_f32_16x16x32_bf16(
                false, a, false, b[t], (short)0, acc[t], false, false);
        if (more) {
            a = aN;
#pragma unroll
            for (int t = 0; t < NT; ++t) b[t] = bN[t];
        }
    }
}

__device__ __forceinline__ void zero_acc(v8f* acc, int nt) {
    for (int t = 0; t < nt; ++t)
#pragma unroll
        for (int v = 0; v < 8; ++v) acc[t][v] = 0.0f;
}

// Online-softmax attention for one (batch row, head): q from LDS (bf16),
// K/V rows from bf16 global caches (pos stride = D_MODEL). Writes 64 bf16 ctx.
__device__ __forceinline__ void attn_head(const __bf16* qv,
                                          const __bf16* __restrict__ Kb,
                                          const __bf16* __restrict__ Vb,
                                          int pmax, __bf16* ctxOut) {
    float mx = -1e30f, l = 0.0f;
    float ctx[DHEAD];
#pragma unroll
    for (int d = 0; d < DHEAD; ++d) ctx[d] = 0.0f;
    for (int p = 0; p < pmax; ++p) {
        const __bf16* kr = Kb + (size_t)p * D_MODEL;
        const __bf16* vr = Vb + (size_t)p * D_MODEL;
        float s = 0.0f;
#pragma unroll
        for (int d = 0; d < DHEAD; d += 8) {
            v8bf qq = *(const v8bf*)(qv + d);
            v8bf kk = *(const v8bf*)(kr + d);
#pragma unroll
            for (int e = 0; e < 8; ++e) s += (float)qq[e] * (float)kk[e];
        }
        s *= 0.125f;                    // 1/sqrt(DHEAD)
        float nm   = fmaxf(mx, s);
        float corr = __expf(mx - nm);
        float w    = __expf(s - nm);
        l = l * corr + w;
#pragma unroll
        for (int d = 0; d < DHEAD; d += 8) {
            v8bf vv = *(const v8bf*)(vr + d);
#pragma unroll
            for (int e = 0; e < 8; ++e) ctx[d + e] = ctx[d + e] * corr + w * (float)vv[e];
        }
        mx = nm;
    }
    float inv = 1.0f / l;
#pragma unroll
    for (int d = 0; d < DHEAD; ++d) ctxOut[d] = (__bf16)(ctx[d] * inv);
}

// Row LayerNorm over 16x512 tile in sX; writes normalized fp32 back to sX and
// bf16 to sAbf. scr: >= 544 floats of scratch. Includes its own barriers.
__device__ __forceinline__ void block_ln(float (*sX)[D_MODEL], __bf16 (*sAbf)[D_MODEL],
                                         float* scr, const float* __restrict__ g,
                                         const float* __restrict__ bta, int tid) {
    __syncthreads();
    int r  = tid >> 4;
    int c0 = (tid & 15) * 32;
    float s = 0.0f, ss = 0.0f;
    for (int c = c0; c < c0 + 32; ++c) { float x = sX[r][c]; s += x; ss += x * x; }
    scr[r * 16 + (tid & 15)]       = s;
    scr[256 + r * 16 + (tid & 15)] = ss;
    __syncthreads();
    if ((tid & 15) == 0) {
        float S = 0.0f, SS = 0.0f;
        for (int j = 0; j < 16; ++j) { S += scr[r * 16 + j]; SS += scr[256 + r * 16 + j]; }
        float mu  = S * (1.0f / 512.0f);
        float var = SS * (1.0f / 512.0f) - mu * mu;
        scr[512 + r] = mu;
        scr[528 + r] = rsqrtf(var + 1e-5f);
    }
    __syncthreads();
    float mu = scr[512 + r], rs = scr[528 + r];
    for (int c = c0; c < c0 + 32; ++c) {
        float y = (sX[r][c] - mu) * rs * g[c] + bta[c];
        sX[r][c]   = y;
        sAbf[r][c] = (__bf16)y;
    }
    __syncthreads();
}

// ---------------------------------------------------------------------------
// One-time cross-attn KV cache: K2/V2 = h_C @ m2_{k,v}w + bias, stored bf16.
// Grid: (B*L_C)/16 WGs of 256 threads; 8 waves cover N=512 (64 cols each).
// ---------------------------------------------------------------------------
__global__ __launch_bounds__(NTHREADS) void kv_cross_kernel(
    const float* __restrict__ hC,
    const __bf16* __restrict__ pkw, const __bf16* __restrict__ pvw,
    const float* __restrict__ kb,  const float* __restrict__ vb,
    __bf16* __restrict__ K2, __bf16* __restrict__ V2) {
    __shared__ float  sF[MT][D_MODEL];
    __shared__ __bf16 sA[MT][D_MODEL];
    const int tid  = threadIdx.x;
    const int lane = tid & 31;
    const int wv   = tid >> 5;
    const size_t r0 = (size_t)blockIdx.x * MT;

    // async DMA the f32 tile into LDS, then convert to bf16
    async_tile_load_f32(&sF[0][0], hC + r0 * D_MODEL, MT * D_MODEL, tid);
    __syncthreads();
    for (int idx = tid; idx < MT * D_MODEL; idx += NTHREADS) {
        int r = idx >> 9, c = idx & 511;
        sA[r][c] = (__bf16)sF[r][c];
    }
    __syncthreads();

    v8f acc[4];
    // K projection
    zero_acc(acc, 4);
    wave_gemm<16, 4>(&sA[0][0], D_MODEL, pkw, wv * 4, acc);
#pragma unroll
    for (int t = 0; t < 4; ++t) {
        int n = (wv * 4 + t) * 16 + (lane & 15);
        float b = kb[n];
#pragma unroll
        for (int v = 0; v < 8; ++v) {
            int m = v + ((lane >> 4) << 3);
            K2[(r0 + m) * D_MODEL + n] = (__bf16)(acc[t][v] + b);
        }
    }
    // V projection
    zero_acc(acc, 4);
    wave_gemm<16, 4>(&sA[0][0], D_MODEL, pvw, wv * 4, acc);
#pragma unroll
    for (int t = 0; t < 4; ++t) {
        int n = (wv * 4 + t) * 16 + (lane & 15);
        float b = vb[n];
#pragma unroll
        for (int v = 0; v < 8; ++v) {
            int m = v + ((lane >> 4) << 3);
            V2[(r0 + m) * D_MODEL + n] = (__bf16)(acc[t][v] + b);
        }
    }
}

struct StepP {
    float* xBuf;
    __bf16 *K1, *V1;
    const __bf16 *K2, *V2;
    const __bf16 *pqw1, *pkw1, *pvw1, *pow1, *pqw2, *pow2, *pfc1, *pfc3;
    const float *qb1, *kb1, *vb1, *ob1, *qb2, *ob2;
    const float *ln1g, *ln1b, *ln2g, *ln2b;
    const float *fc1b, *fc3b, *eps, *pe;
    float *ow, *oz, *om, *olv;
};

// ---------------------------------------------------------------------------
// One decode step, fully fused. Grid: 32 WGs (16 batch rows each) x 256 thr.
// ---------------------------------------------------------------------------
__global__ __launch_bounds__(NTHREADS) void step_kernel(int i, StepP P) {
    __shared__ float  sX[MT][D_MODEL];     // fp32 residual stream       (32 KB)
    __shared__ __bf16 sAbf[MT][D_MODEL];   // bf16 GEMM-A staging        (16 KB)
    __shared__ __bf16 sQ[MT][D_MODEL];     // bf16 q / overlaid scratch  (16 KB)
    float* sScr = (float*)&sQ[0][0];       // LN scratch  (floats [0..543])
    float* sML  = sScr + 1024;             // fc1 output 16x128 fp32 (8 KB region)

    const int tid  = threadIdx.x;
    const int lane = tid & 31;
    const int wv   = tid >> 5;
    const int bt   = blockIdx.x * MT;
    v8f acc[4];

    // phase 0: async DMA token tile x_i straight into sX, then bf16 convert
    async_tile_load_f32(&sX[0][0], P.xBuf + (size_t)bt * D_MODEL, MT * D_MODEL, tid);
    __syncthreads();
    for (int idx = tid; idx < MT * D_MODEL; idx += NTHREADS) {
        int r = idx >> 9, c = idx & 511;
        sAbf[r][c] = (__bf16)sX[r][c];
    }
    __syncthreads();

    // phase 1: q/k/v projections; k,v appended to caches at position i
    zero_acc(acc, 4);
    wave_gemm<16, 4>(&sAbf[0][0], D_MODEL, P.pqw1, wv * 4, acc);
#pragma unroll
    for (int t = 0; t < 4; ++t) {
        int n = (wv * 4 + t) * 16 + (lane & 15);
        float b = P.qb1[n];
#pragma unroll
        for (int v = 0; v < 8; ++v) {
            int m = v + ((lane >> 4) << 3);
            sQ[m][n] = (__bf16)(acc[t][v] + b);
        }
    }
    zero_acc(acc, 4);
    wave_gemm<16, 4>(&sAbf[0][0], D_MODEL, P.pkw1, wv * 4, acc);
#pragma unroll
    for (int t = 0; t < 4; ++t) {
        int n = (wv * 4 + t) * 16 + (lane & 15);
        float b = P.kb1[n];
#pragma unroll
        for (int v = 0; v < 8; ++v) {
            int m = v + ((lane >> 4) << 3);
            P.K1[((size_t)(bt + m) * TT + i) * D_MODEL + n] = (__bf16)(acc[t][v] + b);
        }
    }
    zero_acc(acc, 4);
    wave_gemm<16, 4>(&sAbf[0][0], D_MODEL, P.pvw1, wv * 4, acc);
#pragma unroll
    for (int t = 0; t < 4; ++t) {
        int n = (wv * 4 + t) * 16 + (lane & 15);
        float b = P.vb1[n];
#pragma unroll
        for (int v = 0; v < 8; ++v) {
            int m = v + ((lane >> 4) << 3);
            P.V1[((size_t)(bt + m) * TT + i) * D_MODEL + n] = (__bf16)(acc[t][v] + b);
        }
    }
    __threadfence_block();
    __syncthreads();

    // phase 2: causal self-attention (positions 0..i), 16 rows x 8 heads
    if (tid < MT * NHEADS) {
        int bl = tid >> 3, h = tid & 7;
        const __bf16* Kb = P.K1 + ((size_t)(bt + bl) * TT) * D_MODEL + h * DHEAD;
        const __bf16* Vb = P.V1 + ((size_t)(bt + bl) * TT) * D_MODEL + h * DHEAD;
        attn_head(&sQ[bl][h * DHEAD], Kb, Vb, i + 1, &sAbf[bl][h * DHEAD]);
    }
    __syncthreads();

    // phase 3: o-projection + residual(x) -> LN1 -> w_bar
    zero_acc(acc, 4);
    wave_gemm<16, 4>(&sAbf[0][0], D_MODEL, P.pow1, wv * 4, acc);
#pragma unroll
    for (int t = 0; t < 4; ++t) {
        int n = (wv * 4 + t) * 16 + (lane & 15);
        float b = P.ob1[n];
#pragma unroll
        for (int v = 0; v < 8; ++v) {
            int m = v + ((lane >> 4) << 3);
            sX[m][n] += acc[t][v] + b;
        }
    }
    block_ln(sX, sAbf, sScr, P.ln1g, P.ln1b, tid);

    // phase 4: q2 projection of w_bar
    zero_acc(acc, 4);
    wave_gemm<16, 4>(&sAbf[0][0], D_MODEL, P.pqw2, wv * 4, acc);
#pragma unroll
    for (int t = 0; t < 4; ++t) {
        int n = (wv * 4 + t) * 16 + (lane & 15);
        float b = P.qb2[n];
#pragma unroll
        for (int v = 0; v < 8; ++v) {
            int m = v + ((lane >> 4) << 3);
            sQ[m][n] = (__bf16)(acc[t][v] + b);
        }
    }
    __syncthreads();

    // phase 5: cross-attention over h_C caches (128 positions, L2-resident)
    if (tid < MT * NHEADS) {
        int bl = tid >> 3, h = tid & 7;
        const __bf16* Kb = P.K2 + ((size_t)(bt + bl) * LC) * D_MODEL + h * DHEAD;
        const __bf16* Vb = P.V2 + ((size_t)(bt + bl) * LC) * D_MODEL + h * DHEAD;
        attn_head(&sQ[bl][h * DHEAD], Kb, Vb, LC, &sAbf[bl][h * DHEAD]);
    }
    __syncthreads();

    // phase 6: o2-projection + residual(w_bar) -> LN2 -> w_hat
    zero_acc(acc, 4);
    wave_gemm<16, 4>(&sAbf[0][0], D_MODEL, P.pow2, wv * 4, acc);
#pragma unroll
    for (int t = 0; t < 4; ++t) {
        int n = (wv * 4 + t) * 16 + (lane & 15);
        float b = P.ob2[n];
#pragma unroll
        for (int v = 0; v < 8; ++v) {
            int m = v + ((lane >> 4) << 3);
            sX[m][n] += acc[t][v] + b;
        }
    }
    block_ln(sX, sAbf, sScr, P.ln2g, P.ln2b, tid);

    // phase 7: fc1 (512 -> 128); waves 0..1 cover the 128 cols
    if (wv < 2) {
        zero_acc(acc, 4);
        wave_gemm<16, 4>(&sAbf[0][0], D_MODEL, P.pfc1, wv * 4, acc);
#pragma unroll
        for (int t = 0; t < 4; ++t) {
            int n = (wv * 4 + t) * 16 + (lane & 15);
            float b = P.fc1b[n];
#pragma unroll
            for (int v = 0; v < 8; ++v) {
                int m = v + ((lane >> 4) << 3);
                sML[m * 128 + n] = acc[t][v] + b;
            }
        }
    }
    __syncthreads();

    // phase 7b: z = mean + exp(0.5*logvar)*eps ; emit z/m/lv outputs
    for (int idx = tid; idx < MT * DLAT; idx += NTHREADS) {
        int r = idx >> 6, d = idx & 63;
        float mean = sML[r * 128 + d];
        float lv   = sML[r * 128 + 64 + d];
        float e    = P.eps[((size_t)i * BB + (bt + r)) * DLAT + d];
        float z    = mean + __expf(0.5f * lv) * e;
        size_t ob  = ((size_t)(bt + r) * TT + i) * DLAT + d;
        P.om[ob]  = mean;
        P.olv[ob] = lv;
        P.oz[ob]  = z;
        sAbf[r][d] = (__bf16)z;        // A for fc3 (cols 0..63, stride D_MODEL)
    }
    __syncthreads();

    // phase 8: fc3 (64 -> 512) + w_hat + bias + pe[i] -> w_new; write back
    zero_acc(acc, 4);
    wave_gemm<2, 4>(&sAbf[0][0], D_MODEL, P.pfc3, wv * 4, acc);
#pragma unroll
    for (int t = 0; t < 4; ++t) {
        int n = (wv * 4 + t) * 16 + (lane & 15);
        float add = P.fc3b[n] + P.pe[(size_t)i * D_MODEL + n];
#pragma unroll
        for (int v = 0; v < 8; ++v) {
            int m  = v + ((lane >> 4) << 3);
            float wn = sX[m][n] + acc[t][v] + add;
            P.xBuf[(size_t)(bt + m) * D_MODEL + n] = wn;
            P.ow[((size_t)(bt + m) * TT + i) * D_MODEL + n] = wn;
        }
    }
}

// ---------------------------------------------------------------------------
extern "C" void kernel_launch(void* const* d_in, const int* in_sizes, int n_in,
                              void* d_out, int out_size, void* d_ws, size_t ws_size,
                              hipStream_t stream) {
    (void)in_sizes; (void)n_in; (void)out_size; (void)ws_size;
    const float* h_C   = (const float*)d_in[0];
    const float* w0    = (const float*)d_in[1];
    const float* pe    = (const float*)d_in[2];
    const float* ln1_g = (const float*)d_in[3];
    const float* ln1_b = (const float*)d_in[4];
    const float* ln2_g = (const float*)d_in[5];
    const float* ln2_b = (const float*)d_in[6];
    const float* fc1_w = (const float*)d_in[7];
    const float* fc1_b = (const float*)d_in[8];
    const float* fc3_w = (const float*)d_in[9];
    const float* fc3_b = (const float*)d_in[10];
    const float* eps   = (const float*)d_in[11];
    const float* m1_qw = (const float*)d_in[12]; const float* m1_qb = (const float*)d_in[13];
    const float* m1_kw = (const float*)d_in[14]; const float* m1_kb = (const float*)d_in[15];
    const float* m1_vw = (const float*)d_in[16]; const float* m1_vb = (const float*)d_in[17];
    const float* m1_ow = (const float*)d_in[18]; const float* m1_ob = (const float*)d_in[19];
    const float* m2_qw = (const float*)d_in[20]; const float* m2_qb = (const float*)d_in[21];
    const float* m2_kw = (const float*)d_in[22]; const float* m2_kb = (const float*)d_in[23];
    const float* m2_vw = (const float*)d_in[24]; const float* m2_vb = (const float*)d_in[25];
    const float* m2_ow = (const float*)d_in[26]; const float* m2_ob = (const float*)d_in[27];

    // Workspace carve-out (~175 MB): x buffer, KV caches, packed bf16 weights.
    char* ws = (char*)d_ws;
    size_t off = 0;
    auto take = [&](size_t bytes) -> char* {
        char* p = ws + off;
        off += (bytes + 255) & ~(size_t)255;
        return p;
    };
    float*  xBuf = (float*) take((size_t)BB * D_MODEL * 4);
    __bf16* K1   = (__bf16*)take((size_t)BB * TT * D_MODEL * 2);
    __bf16* V1   = (__bf16*)take((size_t)BB * TT * D_MODEL * 2);
    __bf16* K2   = (__bf16*)take((size_t)BB * LC * D_MODEL * 2);
    __bf16* V2   = (__bf16*)take((size_t)BB * LC * D_MODEL * 2);
    __bf16* pq1  = (__bf16*)take((size_t)D_MODEL * D_MODEL * 2);
    __bf16* pk1  = (__bf16*)take((size_t)D_MODEL * D_MODEL * 2);
    __bf16* pv1  = (__bf16*)take((size_t)D_MODEL * D_MODEL * 2);
    __bf16* po1  = (__bf16*)take((size_t)D_MODEL * D_MODEL * 2);
    __bf16* pq2  = (__bf16*)take((size_t)D_MODEL * D_MODEL * 2);
    __bf16* pk2  = (__bf16*)take((size_t)D_MODEL * D_MODEL * 2);
    __bf16* pv2  = (__bf16*)take((size_t)D_MODEL * D_MODEL * 2);
    __bf16* po2  = (__bf16*)take((size_t)D_MODEL * D_MODEL * 2);
    __bf16* pf1  = (__bf16*)take((size_t)D_MODEL * 128 * 2);
    __bf16* pf3  = (__bf16*)take((size_t)DLAT * D_MODEL * 2);

    // Pack weights into WMMA B-fragment layout (bf16).
    struct PW { const float* s; __bf16* d; int K, N; };
    const PW pws[10] = {
        {m1_qw, pq1, D_MODEL, D_MODEL}, {m1_kw, pk1, D_MODEL, D_MODEL},
        {m1_vw, pv1, D_MODEL, D_MODEL}, {m1_ow, po1, D_MODEL, D_MODEL},
        {m2_qw, pq2, D_MODEL, D_MODEL}, {m2_kw, pk2, D_MODEL, D_MODEL},
        {m2_vw, pv2, D_MODEL, D_MODEL}, {m2_ow, po2, D_MODEL, D_MODEL},
        {fc1_w, pf1, D_MODEL, 128},     {fc3_w, pf3, DLAT, D_MODEL},
    };
    for (int j = 0; j < 10; ++j) {
        int total = pws[j].K * pws[j].N;
        pack_w_kernel<<<(total + 255) / 256, 256, 0, stream>>>(
            pws[j].s, pws[j].d, pws[j].K / 32, pws[j].N);
    }

    init_x_kernel<<<(BB * D_MODEL + 255) / 256, 256, 0, stream>>>(w0, xBuf);

    // One-time cross-attention KV caches (L2-resident for the 32 steps).
    kv_cross_kernel<<<(BB * LC) / MT, NTHREADS, 0, stream>>>(
        h_C, pk2, pv2, m2_kb, m2_vb, K2, V2);

    float* out   = (float*)d_out;
    float* out_w = out;                                     // (B,T,512)
    float* out_z = out + (size_t)BB * TT * D_MODEL;         // (B,T,64)
    float* out_m = out_z + (size_t)BB * TT * DLAT;
    float* out_l = out_m + (size_t)BB * TT * DLAT;

    StepP P;
    P.xBuf = xBuf; P.K1 = K1; P.V1 = V1; P.K2 = K2; P.V2 = V2;
    P.pqw1 = pq1; P.pkw1 = pk1; P.pvw1 = pv1; P.pow1 = po1;
    P.pqw2 = pq2; P.pow2 = po2; P.pfc1 = pf1; P.pfc3 = pf3;
    P.qb1 = m1_qb; P.kb1 = m1_kb; P.vb1 = m1_vb; P.ob1 = m1_ob;
    P.qb2 = m2_qb; P.ob2 = m2_ob;
    P.ln1g = ln1_g; P.ln1b = ln1_b; P.ln2g = ln2_g; P.ln2b = ln2_b;
    P.fc1b = fc1_b; P.fc3b = fc3_b; P.eps = eps; P.pe = pe;
    P.ow = out_w; P.oz = out_z; P.om = out_m; P.olv = out_l;

    for (int i = 0; i < TT; ++i)
        step_kernel<<<BB / MT, NTHREADS, 0, stream>>>(i, P);
}
